// InternLM2MHADecoderLayer_27882927685978
// MI455X (gfx1250) — matrix operand order repaired and checked by reading, not compiled, not run on hardware
//
#include <hip/hip_runtime.h>
#include <hip/hip_bf16.h>
#include <math.h>

// ---------------------------------------------------------------------------
// InternLM2 decoder layer for MI455X (gfx1250, wave32, WMMA).
// bf16 WMMA (v_wmma_f32_16x16x32_bf16) for every GEMM, f32 accumulate.
// Tile staging uses the CDNA5 Tensor Data Mover (tensor_load_to_lds, one DMA
// per 128x32 tile, tracked by TENSORcnt) with double-buffered LDS so DMA
// overlaps WMMA compute.
// ---------------------------------------------------------------------------

#define S_LEN 2048
#define D_DIM 2048
#define NH    16
#define NKV   8
#define HD    128
#define IM    5632
#define QKV_D ((NH + 2 * NKV) * HD)   // 4096

typedef unsigned short u16;
typedef __bf16 bf16x16 __attribute__((ext_vector_type(16)));
typedef float  f32x8   __attribute__((ext_vector_type(8)));
typedef unsigned int u32x4 __attribute__((ext_vector_type(4)));
typedef unsigned int u32x8 __attribute__((ext_vector_type(8)));

union Frag32B { uint4 u[2]; bf16x16 v; };

static __device__ __forceinline__ u16 f32_to_bf16(float f) {
    unsigned int u = __float_as_uint(f);
    unsigned int r = 0x7FFFu + ((u >> 16) & 1u);
    u += r;
    return (u16)(u >> 16);
}

// Low 32 bits of a flat shared-memory address == LDS byte offset (ISA 10.2:
// LDS aperture truncates addr[31:0]).
static __device__ __forceinline__ unsigned lds_offset(const void* p) {
    return (unsigned)(size_t)p;
}

// ---------------------------------------------------------------------------
// TDM: DMA one 2D bf16 tile (tile_h rows x tile_w elems, row stride `stride`
// elems) from global memory into contiguous LDS at lds_off.
// D# group0: count=1 | lds_addr | global_addr[56:0] | type=2
// D# group1: data_size=1 (2B) | tensor_dim0/1 | tile_dim0/1 | dim0_stride
// Issued as inline asm (builtin arity differs across toolchains).
// ---------------------------------------------------------------------------
static __device__ __forceinline__ void tdm_load_tile_bf16(
    unsigned lds_off, const void* gptr,
    unsigned tile_w, unsigned tile_h, unsigned stride)
{
    unsigned long long ga = (unsigned long long)(size_t)gptr;
    u32x4 g0;
    g0[0] = 0x1u;                                        // count=1, user desc
    g0[1] = lds_off;                                     // lds_addr (bytes)
    g0[2] = (unsigned)(ga & 0xFFFFFFFFu);                // global_addr[31:0]
    g0[3] = (unsigned)((ga >> 32) & 0x01FFFFFFu)         // global_addr[56:32]
          | 0x80000000u;                                 // type=2 (image)
    u32x8 g1;
    g1[0] = 0x00010000u;                                 // data_size=1 -> 2B
    g1[1] = (tile_w & 0xFFFFu) << 16;                    // tensor_dim0[15:0]
    g1[2] = (tile_h & 0xFFFFu) << 16;                    // tensor_dim1[15:0]
    g1[3] = (tile_w & 0xFFFFu) << 16;                    // tile_dim0
    g1[4] = (tile_h & 0xFFFFu);                          // tile_dim1 (dim2=0)
    g1[5] = stride;                                      // dim0_stride[31:0]
    g1[6] = 0u;                                          // stride hi / dim1_stride lo
    g1[7] = 0u;
    asm volatile("tensor_load_to_lds %0, %1" :: "s"(g0), "s"(g1) : "memory");
}

// ---------------------------------------------------------------------------
// Generic tiled GEMM: C[M,N] (f32) = A[M,K] (bf16, row-major, lda) *
//                                    W[N,K]^T (bf16, row-major, ldb)
//                                    (+ residual R[M,N] if R != nullptr)
// Block tile 128x128, K-step 32.  8 wave32 waves, each wave: 4x2 WMMA tiles.
// Tile staging: wave 0 issues one TDM DMA per tile; TENSORcnt + barrier
// publish it.  Double-buffered so the next tiles stream while computing.
// Requires: M,N multiples of 128; K multiple of 32; 16B-aligned pointers.
// ---------------------------------------------------------------------------
__global__ __launch_bounds__(256) void gemm_bf16_wmma(
    const u16* __restrict__ A, int lda,
    const u16* __restrict__ W, int ldb,
    float* __restrict__ C, int ldc,
    const float* __restrict__ R, int ldr,
    int M, int N, int K)
{
    __shared__ u16 As[2][128 * 32];
    __shared__ u16 Bs[2][128 * 32];

    const int t     = threadIdx.x;
    const int lane  = t & 31;
    const int wid   = t >> 5;
    const int lhalf = lane >> 4;   // 0 or 1
    const int l15   = lane & 15;
    const int waveM = wid >> 2;    // 0..1
    const int waveN = wid & 3;     // 0..3
    const int blockM = blockIdx.y * 128;
    const int blockN = blockIdx.x * 128;

    const f32x8 vzero = {0.f, 0.f, 0.f, 0.f, 0.f, 0.f, 0.f, 0.f};
    f32x8 acc[4][2];
    #pragma unroll
    for (int mi = 0; mi < 4; ++mi)
        #pragma unroll
        for (int ni = 0; ni < 2; ++ni)
            acc[mi][ni] = vzero;

    // wave 0 DMAs both 128x32 tiles of buffer `buf` for K-offset k0
    auto fill = [&](int buf, int k0) {
        if (wid == 0) {
            tdm_load_tile_bf16(lds_offset(&As[buf][0]),
                               A + (size_t)blockM * lda + k0, 32, 128,
                               (unsigned)lda);
            tdm_load_tile_bf16(lds_offset(&Bs[buf][0]),
                               W + (size_t)blockN * ldb + k0, 32, 128,
                               (unsigned)ldb);
        }
    };

    // one K-step of WMMA from buffer `buf`
    auto compute = [&](int buf) {
        // A fragments (16x32 bf16): lanes 0-15 hold K 0..15 of row
        // M=l15, lanes 16-31 hold K 16..31.
        Frag32B a[4];
        #pragma unroll
        for (int mi = 0; mi < 4; ++mi) {
            const u16* p = &As[buf][(waveM * 64 + mi * 16 + l15) * 32];
            a[mi].u[0] = *(const uint4*)(p + lhalf * 16);
            a[mi].u[1] = *(const uint4*)(p + lhalf * 16 + 8);
        }
        // B fragments (32x16 bf16): lanes 0-15 hold K 0..15 of col N=l15,
        // lanes 16-31 hold K 16..31.
        Frag32B b[2];
        #pragma unroll
        for (int ni = 0; ni < 2; ++ni) {
            const u16* p = &Bs[buf][(waveN * 32 + ni * 16 + l15) * 32 + lhalf * 16];
            b[ni].u[0] = ((const uint4*)p)[0];
            b[ni].u[1] = ((const uint4*)p)[1];
        }
        #pragma unroll
        for (int mi = 0; mi < 4; ++mi)
            #pragma unroll
            for (int ni = 0; ni < 2; ++ni)
                acc[mi][ni] = __builtin_amdgcn_wmma_f32_16x16x32_bf16(
                    false, a[mi].v, false, b[ni].v,
                    (short)0, acc[mi][ni], false, false);
    };

    fill(0, 0);
    int cur = 0;

    // steady state: next tiles in flight while current buffer computes
    for (int k0 = 0; k0 + 32 < K; k0 += 32) {
        fill(cur ^ 1, k0 + 32);
        __builtin_amdgcn_s_wait_tensorcnt(2);   // current buffer's 2 DMAs done
        __syncthreads();                        // publish to all waves
        compute(cur);
        __syncthreads();                        // done reading buf `cur`
        cur ^= 1;
    }
    // peeled tail
    __builtin_amdgcn_s_wait_tensorcnt(0);
    __syncthreads();
    compute(cur);

    // C layout: VGPR r -> M = r + 8*lhalf, N = l15
    #pragma unroll
    for (int mi = 0; mi < 4; ++mi)
        #pragma unroll
        for (int ni = 0; ni < 2; ++ni)
            #pragma unroll
            for (int r = 0; r < 8; ++r) {
                int row = blockM + waveM * 64 + mi * 16 + lhalf * 8 + r;
                int col = blockN + waveN * 32 + ni * 16 + l15;
                float v = acc[mi][ni][r];
                if (R) v += R[(size_t)row * ldr + col];
                C[(size_t)row * ldc + col] = v;
            }
}

// ---------------------------------------------------------------------------
// RMSNorm over rows of [rows, ncols] f32 -> bf16
// ---------------------------------------------------------------------------
__global__ __launch_bounds__(256) void rmsnorm_bf16_kernel(
    const float* __restrict__ x, const float* __restrict__ w,
    u16* __restrict__ out, int ncols)
{
    __shared__ float red[256];
    const int row = blockIdx.x;
    const float* xr = x + (size_t)row * ncols;
    float s = 0.f;
    for (int c = threadIdx.x; c < ncols; c += 256) {
        float v = xr[c];
        s += v * v;
    }
    red[threadIdx.x] = s;
    __syncthreads();
    for (int off = 128; off > 0; off >>= 1) {
        if (threadIdx.x < off) red[threadIdx.x] += red[threadIdx.x + off];
        __syncthreads();
    }
    float rinv = rsqrtf(red[0] / (float)ncols + 1e-5f);
    for (int c = threadIdx.x; c < ncols; c += 256)
        out[(size_t)row * ncols + c] = f32_to_bf16(xr[c] * rinv * w[c]);
}

// ---------------------------------------------------------------------------
// RoPE + scatter qkv f32 [S, 4096] -> q bf16 [S, NH*HD], k bf16 [S, NKV*HD],
// vT bf16 [NKV*HD, S]  (V transposed so P@V reuses the W[N,K] GEMM operand)
// ---------------------------------------------------------------------------
__global__ __launch_bounds__(256) void rope_scatter_kernel(
    const float* __restrict__ qkv,
    u16* __restrict__ q, u16* __restrict__ k, u16* __restrict__ vT)
{
    const int s = blockIdx.x;
    const float pos = (float)s;
    const float nlb = -logf(10000.0f) / 64.0f;   // 2j/HD exponent scale
    const float* row = qkv + (size_t)s * QKV_D;

    // q heads: NH * 64 rotation pairs
    for (int idx = threadIdx.x; idx < NH * 64; idx += 256) {
        int h = idx >> 6, j = idx & 63;
        int base = (h >> 1) * 512 + (h & 1) * 128;
        float inv = expf(nlb * (float)j);
        float cv = cosf(pos * inv), sv = sinf(pos * inv);
        float t1 = row[base + j], t2 = row[base + j + 64];
        q[(size_t)s * (NH * HD) + h * 128 + j]      = f32_to_bf16(t1 * cv - t2 * sv);
        q[(size_t)s * (NH * HD) + h * 128 + j + 64] = f32_to_bf16(t2 * cv + t1 * sv);
    }
    // k heads: NKV * 64 rotation pairs
    for (int idx = threadIdx.x; idx < NKV * 64; idx += 256) {
        int kv = idx >> 6, j = idx & 63;
        int base = kv * 512 + 256;
        float inv = expf(nlb * (float)j);
        float cv = cosf(pos * inv), sv = sinf(pos * inv);
        float t1 = row[base + j], t2 = row[base + j + 64];
        k[(size_t)s * (NKV * HD) + kv * 128 + j]      = f32_to_bf16(t1 * cv - t2 * sv);
        k[(size_t)s * (NKV * HD) + kv * 128 + j + 64] = f32_to_bf16(t2 * cv + t1 * sv);
    }
    // v: transpose into [NKV*HD, S]
    for (int idx = threadIdx.x; idx < NKV * HD; idx += 256) {
        int kv = idx >> 7, d = idx & 127;
        vT[(size_t)(kv * 128 + d) * S_LEN + s] =
            f32_to_bf16(row[kv * 512 + 384 + d]);
    }
}

// ---------------------------------------------------------------------------
// Causal softmax on scores [S,S] f32 -> P [S,S] bf16, with scale
// ---------------------------------------------------------------------------
__global__ __launch_bounds__(256) void softmax_causal_kernel(
    const float* __restrict__ Sc, u16* __restrict__ P, float scale)
{
    __shared__ float red[256];
    const int row = blockIdx.x;
    const float* sr = Sc + (size_t)row * S_LEN;
    const int n = row + 1;

    float m = -3.0e38f;
    for (int c = threadIdx.x; c < n; c += 256) m = fmaxf(m, sr[c] * scale);
    red[threadIdx.x] = m;
    __syncthreads();
    for (int off = 128; off > 0; off >>= 1) {
        if (threadIdx.x < off)
            red[threadIdx.x] = fmaxf(red[threadIdx.x], red[threadIdx.x + off]);
        __syncthreads();
    }
    m = red[0];
    __syncthreads();

    float sum = 0.f;
    for (int c = threadIdx.x; c < n; c += 256) sum += expf(sr[c] * scale - m);
    red[threadIdx.x] = sum;
    __syncthreads();
    for (int off = 128; off > 0; off >>= 1) {
        if (threadIdx.x < off) red[threadIdx.x] += red[threadIdx.x + off];
        __syncthreads();
    }
    float inv = 1.0f / red[0];

    for (int c = threadIdx.x; c < S_LEN; c += 256) {
        float p = (c < n) ? expf(sr[c] * scale - m) * inv : 0.f;
        P[(size_t)row * S_LEN + c] = f32_to_bf16(p);
    }
}

// ---------------------------------------------------------------------------
// Elementwise helpers
// ---------------------------------------------------------------------------
__global__ __launch_bounds__(256) void cvt_f32_bf16_kernel(
    const float* __restrict__ in, u16* __restrict__ out, size_t n)
{
    size_t i = (size_t)blockIdx.x * blockDim.x + threadIdx.x;
    size_t stride = (size_t)gridDim.x * blockDim.x;
    for (; i < n; i += stride) out[i] = f32_to_bf16(in[i]);
}

__global__ __launch_bounds__(256) void silu_gate_bf16_kernel(
    const float* __restrict__ g1, const float* __restrict__ g3,
    u16* __restrict__ out, size_t n)
{
    size_t i = (size_t)blockIdx.x * blockDim.x + threadIdx.x;
    size_t stride = (size_t)gridDim.x * blockDim.x;
    for (; i < n; i += stride) {
        float a = g1[i];
        float sil = a / (1.0f + expf(-a));
        out[i] = f32_to_bf16(sil * g3[i]);
    }
}

// ---------------------------------------------------------------------------
// Host-side orchestration
// ---------------------------------------------------------------------------
extern "C" void kernel_launch(void* const* d_in, const int* in_sizes, int n_in,
                              void* d_out, int out_size, void* d_ws, size_t ws_size,
                              hipStream_t stream)
{
    (void)in_sizes; (void)n_in; (void)out_size; (void)ws_size;

    const float* x      = (const float*)d_in[0];
    const float* w_qkv  = (const float*)d_in[1];
    const float* w_out  = (const float*)d_in[2];
    const float* w1     = (const float*)d_in[3];
    const float* w2     = (const float*)d_in[4];
    const float* w3     = (const float*)d_in[5];
    const float* anw    = (const float*)d_in[6];
    const float* fnw    = (const float*)d_in[7];
    float* out          = (float*)d_out;

    // workspace carve-up (256B aligned)
    char* base = (char*)d_ws;
    size_t off = 0;
    auto alloc = [&](size_t bytes) -> void* {
        void* p = base + off;
        off += (bytes + 255) & ~(size_t)255;
        return p;
    };

    u16* w_qkv_b = (u16*)alloc((size_t)QKV_D * D_DIM * 2);
    u16* w_out_b = (u16*)alloc((size_t)D_DIM * (NH * HD) * 2);
    u16* w1_b    = (u16*)alloc((size_t)IM * D_DIM * 2);
    u16* w3_b    = (u16*)alloc((size_t)IM * D_DIM * 2);
    u16* w2_b    = (u16*)alloc((size_t)D_DIM * IM * 2);
    u16* xh_b    = (u16*)alloc((size_t)S_LEN * D_DIM * 2);
    float* qkv   = (float*)alloc((size_t)S_LEN * QKV_D * 4);
    u16* q_b     = (u16*)alloc((size_t)S_LEN * NH * HD * 2);
    u16* k_b     = (u16*)alloc((size_t)S_LEN * NKV * HD * 2);
    u16* vT_b    = (u16*)alloc((size_t)NKV * HD * S_LEN * 2);
    float* sc    = (float*)alloc((size_t)S_LEN * S_LEN * 4);
    u16* p_b     = (u16*)alloc((size_t)S_LEN * S_LEN * 2);
    float* ctx   = (float*)alloc((size_t)S_LEN * NH * HD * 4);
    u16* ctx_b   = (u16*)alloc((size_t)S_LEN * NH * HD * 2);
    float* h     = (float*)alloc((size_t)S_LEN * D_DIM * 4);
    u16* h2_b    = (u16*)alloc((size_t)S_LEN * D_DIM * 2);
    float* g1    = (float*)alloc((size_t)S_LEN * IM * 4);
    float* g3    = (float*)alloc((size_t)S_LEN * IM * 4);
    u16* hup_b   = (u16*)alloc((size_t)S_LEN * IM * 2);

    const dim3 blk(256);
    const int CVT_GRID = 2048;

    // 1) weights f32 -> bf16
    cvt_f32_bf16_kernel<<<CVT_GRID, blk, 0, stream>>>(w_qkv, w_qkv_b, (size_t)QKV_D * D_DIM);
    cvt_f32_bf16_kernel<<<CVT_GRID, blk, 0, stream>>>(w_out, w_out_b, (size_t)D_DIM * NH * HD);
    cvt_f32_bf16_kernel<<<CVT_GRID, blk, 0, stream>>>(w1, w1_b, (size_t)IM * D_DIM);
    cvt_f32_bf16_kernel<<<CVT_GRID, blk, 0, stream>>>(w3, w3_b, (size_t)IM * D_DIM);
    cvt_f32_bf16_kernel<<<CVT_GRID, blk, 0, stream>>>(w2, w2_b, (size_t)D_DIM * IM);

    // 2) attn RMSNorm
    rmsnorm_bf16_kernel<<<S_LEN, blk, 0, stream>>>(x, anw, xh_b, D_DIM);

    // 3) qkv = xh @ w_qkv^T   [2048 x 4096]
    gemm_bf16_wmma<<<dim3(QKV_D / 128, S_LEN / 128), blk, 0, stream>>>(
        xh_b, D_DIM, w_qkv_b, D_DIM, qkv, QKV_D, nullptr, 0,
        S_LEN, QKV_D, D_DIM);

    // 4) RoPE + scatter to q/k/vT
    rope_scatter_kernel<<<S_LEN, blk, 0, stream>>>(qkv, q_b, k_b, vT_b);

    // 5) attention per head
    const float scale = 0.08838834764831845f;   // 1/sqrt(128)
    for (int hh = 0; hh < NH; ++hh) {
        int kv = hh >> 1;
        // scores = q_h @ k_h^T   [2048 x 2048], K = 128
        gemm_bf16_wmma<<<dim3(S_LEN / 128, S_LEN / 128), blk, 0, stream>>>(
            q_b + hh * HD, NH * HD, k_b + kv * HD, NKV * HD,
            sc, S_LEN, nullptr, 0, S_LEN, S_LEN, HD);
        // causal softmax -> bf16 P
        softmax_causal_kernel<<<S_LEN, blk, 0, stream>>>(sc, p_b, scale);
        // ctx_h = P @ V_h      (vT is [HD, S] => exactly W[N,K] operand)
        gemm_bf16_wmma<<<dim3(HD / 128, S_LEN / 128), blk, 0, stream>>>(
            p_b, S_LEN, vT_b + (size_t)kv * HD * S_LEN, S_LEN,
            ctx + hh * HD, NH * HD, nullptr, 0, S_LEN, HD, S_LEN);
    }

    // 6) ctx f32 -> bf16
    cvt_f32_bf16_kernel<<<CVT_GRID, blk, 0, stream>>>(ctx, ctx_b, (size_t)S_LEN * NH * HD);

    // 7) h = x + ctx @ w_out^T
    gemm_bf16_wmma<<<dim3(D_DIM / 128, S_LEN / 128), blk, 0, stream>>>(
        ctx_b, NH * HD, w_out_b, NH * HD, h, D_DIM, x, D_DIM,
        S_LEN, D_DIM, NH * HD);

    // 8) ffn RMSNorm
    rmsnorm_bf16_kernel<<<S_LEN, blk, 0, stream>>>(h, fnw, h2_b, D_DIM);

    // 9) g1 = h2 @ w1^T ; g3 = h2 @ w3^T   [2048 x 5632]
    gemm_bf16_wmma<<<dim3(IM / 128, S_LEN / 128), blk, 0, stream>>>(
        h2_b, D_DIM, w1_b, D_DIM, g1, IM, nullptr, 0, S_LEN, IM, D_DIM);
    gemm_bf16_wmma<<<dim3(IM / 128, S_LEN / 128), blk, 0, stream>>>(
        h2_b, D_DIM, w3_b, D_DIM, g3, IM, nullptr, 0, S_LEN, IM, D_DIM);

    // 10) hup = silu(g1) * g3 -> bf16
    silu_gate_bf16_kernel<<<CVT_GRID, blk, 0, stream>>>(g1, g3, hup_b, (size_t)S_LEN * IM);

    // 11) out = h + hup @ w2^T
    gemm_bf16_wmma<<<dim3(D_DIM / 128, S_LEN / 128), blk, 0, stream>>>(
        hup_b, IM, w2_b, IM, out, D_DIM, h, D_DIM,
        S_LEN, D_DIM, IM);
}
